// CausalSelfAttention_9466107921065
// MI455X (gfx1250) — compile-verified, matrix operand order checked by reading, not run on hardware
//
#include <hip/hip_runtime.h>

#define BB 4
#define SS 2048
#define EE 1024
#define HH 16
#define DD 64

typedef __attribute__((ext_vector_type(16))) __bf16 v16bf;
typedef __attribute__((ext_vector_type(8)))  __bf16 v8bf;
typedef __attribute__((ext_vector_type(4)))  __bf16 v4bf;
typedef __attribute__((ext_vector_type(8)))  float  v8f;

// Load a 16-element bf16 WMMA fragment as two contiguous 16-byte chunks.
// Fragment layout (A and B identical on CDNA5): lane L -> row/col = L%16,
// K-chunks: lanes 0-15 hold K [k0..k0+7] and [k0+16..k0+23],
//           lanes 16-31 hold K [k0+8..k0+15] and [k0+24..k0+31].
static __device__ __forceinline__ v16bf load_frag(const __bf16* lo, const __bf16* hi) {
  v8bf a = *(const v8bf*)lo;
  v8bf b = *(const v8bf*)hi;
  v16bf r;
#pragma unroll
  for (int i = 0; i < 8; ++i) { r[i] = a[i]; r[i + 8] = b[i]; }
  return r;
}

static __device__ __forceinline__ v8f wmma_bf16(v16bf a, v16bf b, v8f c) {
  // (neg_a, A, neg_b, B, c_mod, C, reuse_a, reuse_b)
  return __builtin_amdgcn_wmma_f32_16x16x32_bf16(false, a, false, b, (short)0, c,
                                                 false, false);
}

// ---------------- elementwise f32 -> bf16 (vectorized x4) ----------------
__global__ void __launch_bounds__(256) cvt_bf16(const float* __restrict__ s,
                                                __bf16* __restrict__ d, int n4) {
  int i = blockIdx.x * blockDim.x + threadIdx.x;
  if (i >= n4) return;
  float4 v = ((const float4*)s)[i];
  v4bf o;
  o[0] = (__bf16)v.x; o[1] = (__bf16)v.y; o[2] = (__bf16)v.z; o[3] = (__bf16)v.w;
  ((v4bf*)d)[i] = o;
}

// ---------------- transpose + convert: dst[c][r] = (bf16)src[r][c] ----------------
__global__ void __launch_bounds__(256) xpose_bf16(const float* __restrict__ s,
                                                  __bf16* __restrict__ d, int R, int C) {
  __shared__ float t[32][33];
  int c0 = blockIdx.x * 32, r0 = blockIdx.y * 32;
  int tx = threadIdx.x & 31, ty = threadIdx.x >> 5;
#pragma unroll
  for (int i = ty; i < 32; i += 8) t[i][tx] = s[(size_t)(r0 + i) * C + c0 + tx];
  __syncthreads();
#pragma unroll
  for (int i = ty; i < 32; i += 8)
    d[(size_t)(c0 + i) * R + r0 + tx] = (__bf16)t[tx][i];
}

// ---------------- bf16 GEMM, 64x32 tile per wave, k-step 32 ----------------
// MODE 0: C = A*B + bqkv, scatter to Q[B,H,S,D] (pre-scaled by 1/sqrt(D)),
//         K[B,H,S,D], Vt[B,H,D,S] (bf16)
// MODE 1: C = A*B + bo  -> f32 Out[M,N]
template <int MODE>
__global__ void __launch_bounds__(256)
gemm_bf16(const __bf16* __restrict__ A, const __bf16* __restrict__ Bt,
          const float* __restrict__ bias,
          __bf16* __restrict__ Qb, __bf16* __restrict__ Kb,
          __bf16* __restrict__ Vt, float* __restrict__ Out, int Nc, int Kd) {
  const int lane = threadIdx.x & 31;
  const int l16 = lane & 15, lh = lane >> 4;
  const int wid = blockIdx.x * (blockDim.x >> 5) + (threadIdx.x >> 5);
  const int wnT = Nc >> 5;
  const int wm = wid / wnT, wn = wid % wnT;

  const __bf16* Ar[4];
  const __bf16* Br[2];
#pragma unroll
  for (int mt = 0; mt < 4; ++mt) Ar[mt] = A + (size_t)(wm * 64 + mt * 16 + l16) * Kd;
#pragma unroll
  for (int nt = 0; nt < 2; ++nt) Br[nt] = Bt + (size_t)(wn * 32 + nt * 16 + l16) * Kd;

  v8f acc[4][2] = {};
  for (int k0 = 0; k0 < Kd; k0 += 32) {
    const int klo = k0 + lh * 8;
    v16bf a[4], b[2];
#pragma unroll
    for (int mt = 0; mt < 4; ++mt) a[mt] = load_frag(Ar[mt] + klo, Ar[mt] + klo + 16);
#pragma unroll
    for (int nt = 0; nt < 2; ++nt) b[nt] = load_frag(Br[nt] + klo, Br[nt] + klo + 16);
#pragma unroll
    for (int mt = 0; mt < 4; ++mt)
#pragma unroll
      for (int nt = 0; nt < 2; ++nt)
        acc[mt][nt] = wmma_bf16(a[mt], b[nt], acc[mt][nt]);
  }

#pragma unroll
  for (int mt = 0; mt < 4; ++mt) {
#pragma unroll
    for (int nt = 0; nt < 2; ++nt) {
      const int col = wn * 32 + nt * 16 + l16;
      const float bv = bias[col];
#pragma unroll
      for (int j = 0; j < 8; ++j) {
        const int row = wm * 64 + mt * 16 + j + lh * 8;  // C layout: M=j+8*(lane/16)
        const float v = acc[mt][nt][j] + bv;
        if (MODE == 0) {
          const int which = col >> 10;           // col / E
          const int e = col & (EE - 1);
          const int h = e >> 6, dd = e & 63;
          const int bb = row >> 11, ss = row & (SS - 1);
          const int bh = bb * HH + h;
          if (which == 0)      Qb[((size_t)bh * SS + ss) * DD + dd] = (__bf16)(v * 0.125f);
          else if (which == 1) Kb[((size_t)bh * SS + ss) * DD + dd] = (__bf16)v;
          else                 Vt[((size_t)bh * DD + dd) * SS + ss] = (__bf16)v;
        } else {
          Out[(size_t)row * Nc + col] = v;
        }
      }
    }
  }
}

// ---------------- one 32-key flash-attention step ----------------
// S^T = K * Q^T (C layout: M=key, N=query=lane%16) -> per-lane softmax state,
// P^T C-tiles map directly onto B-fragments of O^T = V^T * P^T.
// MASK=false: step guaranteed fully unmasked (kb+31 < qbase) -> no compares,
//             and prefetch the next step's K rows.
template <bool MASK>
static __device__ __forceinline__ void attn_step(
    const __bf16* __restrict__ Kh, const __bf16* __restrict__ Vh, int kb,
    v16bf qf0, v16bf qf1, int q, int l16, int lh, int klo,
    float& m, float& lsum, v8f o[4]) {
  const __bf16* kr0 = Kh + (size_t)(kb + l16) * DD;
  const __bf16* kr1 = Kh + (size_t)(kb + 16 + l16) * DD;
  v16bf k00 = load_frag(kr0 + klo,      kr0 + klo + 16);
  v16bf k01 = load_frag(kr0 + 32 + klo, kr0 + 48 + klo);
  v16bf k10 = load_frag(kr1 + klo,      kr1 + klo + 16);
  v16bf k11 = load_frag(kr1 + 32 + klo, kr1 + 48 + klo);

  if (!MASK) {
    // Pull next step's K rows toward the caches (global_prefetch_b8).
    __builtin_prefetch(kr0 + 32 * DD, 0, 3);
    __builtin_prefetch(kr1 + 32 * DD, 0, 3);
  }

  v8f s0 = {}, s1 = {};
  s0 = wmma_bf16(k00, qf0, s0);
  s0 = wmma_bf16(k01, qf1, s0);
  s1 = wmma_bf16(k10, qf0, s1);
  s1 = wmma_bf16(k11, qf1, s1);

  if (MASK) {
#pragma unroll
    for (int j = 0; j < 8; ++j) {
      const int key0 = kb + j + lh * 8;            // S^T C layout: M = key
      s0[j] = (key0      <= q) ? s0[j] : -3.0e38f;
      s1[j] = (key0 + 16 <= q) ? s1[j] : -3.0e38f;
    }
  }

  // Balanced-tree row max over the 16 scores held in this lane.
  float mx[4];
#pragma unroll
  for (int j = 0; j < 4; ++j)
    mx[j] = fmaxf(fmaxf(s0[2 * j], s0[2 * j + 1]),
                  fmaxf(s1[2 * j], s1[2 * j + 1]));
  float tmax = fmaxf(fmaxf(mx[0], mx[1]), fmaxf(mx[2], mx[3]));
  tmax = fmaxf(tmax, __shfl_xor(tmax, 16, 32));
  const float nm = fmaxf(m, tmax);
  const float alpha = __expf(m - nm);

  v16bf pf;
  float ps0[8], ps1[8];
#pragma unroll
  for (int j = 0; j < 8; ++j) {
    ps0[j] = __expf(s0[j] - nm);
    ps1[j] = __expf(s1[j] - nm);
    pf[j]     = (__bf16)ps0[j];  // keys 0..7 (lanes<16) / 8..15 (lanes>=16)
    pf[j + 8] = (__bf16)ps1[j];  // keys 16..23 / 24..31  == B-fragment layout
  }
  // Balanced-tree row sum.
  float sm[4];
#pragma unroll
  for (int j = 0; j < 4; ++j)
    sm[j] = (ps0[2 * j] + ps0[2 * j + 1]) + (ps1[2 * j] + ps1[2 * j + 1]);
  float rs = (sm[0] + sm[1]) + (sm[2] + sm[3]);
  rs += __shfl_xor(rs, 16, 32);
  lsum = lsum * alpha + rs;
  m = nm;

#pragma unroll
  for (int t = 0; t < 4; ++t)
#pragma unroll
    for (int j = 0; j < 8; ++j) o[t][j] *= alpha;  // O^T col = query = lane%16

#pragma unroll
  for (int t = 0; t < 4; ++t) {
    const __bf16* vr = Vh + (size_t)(t * 16 + l16) * SS + kb;  // A-frag: M=d
    v16bf vf = load_frag(vr + klo, vr + klo + 16);
    o[t] = wmma_bf16(vf, pf, o[t]);
  }
}

// ---------------- flash attention, one 16-query tile per wave ----------------
__global__ void __launch_bounds__(256)
attn_fwd(const __bf16* __restrict__ Qb, const __bf16* __restrict__ Kb,
         const __bf16* __restrict__ Vt, __bf16* __restrict__ At) {
  const int lane = threadIdx.x & 31;
  const int l16 = lane & 15, lh = lane >> 4;
  const int wid = blockIdx.x * (blockDim.x >> 5) + (threadIdx.x >> 5);
  const int qt = wid & (SS / 16 - 1);
  const int bh = wid >> 7;
  const int qbase = qt * 16;
  const int q = qbase + l16;
  const int klo = lh * 8;

  const __bf16* qrow = Qb + ((size_t)bh * SS + q) * DD;
  v16bf qf0 = load_frag(qrow + klo,      qrow + klo + 16);   // d 0..31
  v16bf qf1 = load_frag(qrow + 32 + klo, qrow + 48 + klo);   // d 32..63

  const __bf16* Kh = Kb + (size_t)bh * SS * DD;
  const __bf16* Vh = Vt + (size_t)bh * DD * SS;

  float m = -3.0e38f, lsum = 0.f;
  v8f o[4] = {};

  // Steps with kb+31 < qbase are fully unmasked; exactly one masked step
  // remains at kb = qbase & ~31 (covers the causal diagonal of this q-tile).
  const int kfull = qbase & ~31;
  for (int kb = 0; kb < kfull; kb += 32)
    attn_step<false>(Kh, Vh, kb, qf0, qf1, q, l16, lh, klo, m, lsum, o);
  attn_step<true>(Kh, Vh, kfull, qf0, qf1, q, l16, lh, klo, m, lsum, o);

  const float linv = 1.0f / lsum;
  const int bb = bh >> 4, h = bh & 15;
  __bf16* ar = At + ((size_t)bb * SS + q) * EE + h * DD;
#pragma unroll
  for (int t = 0; t < 4; ++t)
#pragma unroll
    for (int j = 0; j < 8; ++j)
      ar[t * 16 + j + lh * 8] = (__bf16)(o[t][j] * linv);   // O^T: M = d
}

extern "C" void kernel_launch(void* const* d_in, const int* in_sizes, int n_in,
                              void* d_out, int out_size, void* d_ws, size_t ws_size,
                              hipStream_t stream) {
  (void)in_sizes; (void)n_in; (void)out_size; (void)ws_size;
  const float* x    = (const float*)d_in[0];
  const float* Wqkv = (const float*)d_in[1];
  const float* bqkv = (const float*)d_in[2];
  const float* Wo   = (const float*)d_in[3];
  const float* bo   = (const float*)d_in[4];
  float* out = (float*)d_out;

  const size_t nX = (size_t)BB * SS * EE;  // 8M elements
  // workspace layout (bf16): 88 MB total
  __bf16* Xb    = (__bf16*)d_ws;
  __bf16* WqkvT = Xb + nX;                       // [3E][E]
  __bf16* WoT   = WqkvT + (size_t)3 * EE * EE;   // [E][E]
  __bf16* Qb    = WoT + (size_t)EE * EE;         // [B,H,S,D] (pre-scaled)
  __bf16* Kbf   = Qb + nX;                       // [B,H,S,D]
  __bf16* Vt    = Kbf + nX;                      // [B,H,D,S]
  __bf16* At    = Vt + nX;                       // [B,S,E]

  cvt_bf16<<<(int)(nX / 4 / 256), 256, 0, stream>>>(x, Xb, (int)(nX / 4));
  xpose_bf16<<<dim3(3 * EE / 32, EE / 32), 256, 0, stream>>>(Wqkv, WqkvT, EE, 3 * EE);
  xpose_bf16<<<dim3(EE / 32, EE / 32), 256, 0, stream>>>(Wo, WoT, EE, EE);

  // QKV projection: M=8192, N=3072, K=1024 -> 128*96 wave-tiles / 8 waves per block
  gemm_bf16<0><<<(BB * SS / 64) * (3 * EE / 32) / 8, 256, 0, stream>>>(
      Xb, WqkvT, bqkv, Qb, Kbf, Vt, nullptr, 3 * EE, EE);

  // Attention: B*H*(S/16) = 8192 waves / 8 per block
  attn_fwd<<<(BB * HH * (SS / 16)) / 8, 256, 0, stream>>>(Qb, Kbf, Vt, At);

  // Output projection: M=8192, N=1024, K=1024
  gemm_bf16<1><<<(BB * SS / 64) * (EE / 32) / 8, 256, 0, stream>>>(
      At, WoT, bo, nullptr, nullptr, nullptr, out, EE, EE);
}
